// WeaveModule_10505490006487
// MI455X (gfx1250) — matrix operand order repaired and checked by reading, not compiled, not run on hardware
//
#include <hip/hip_runtime.h>

// ---------------------------------------------------------------------------
// MI455X / gfx1250 Weave-module forward, f16 WMMA (v_wmma_f32_16x16x32_f16).
// Wave32; one wave owns 16x16 output tiles; 256-thread blocks = 8 waves
// covering all C=128 columns.  Weights pre-shuffled into WMMA fragment-major
// layout (one contiguous 32B read per lane per B-fragment).  The pair
// mega-kernel processes 32 rows/block so each B-fragment feeds 2 M-tiles.
// ---------------------------------------------------------------------------

typedef _Float16 f16;
typedef __attribute__((ext_vector_type(4)))  _Float16 v4h;
typedef __attribute__((ext_vector_type(16))) _Float16 v16h;
typedef __attribute__((ext_vector_type(8)))  float    v8f;

#define WMMA(a, b, c) __builtin_amdgcn_wmma_f32_16x16x32_f16( \
    false, (a), false, (b), (short)0, (c), false, false)

// A fragment: 16x32 tile (M x K) from row-major f16 matrix, contiguous layout:
// lanes 0-15 : row M=lane,    K 0-15 in v0-7
// lanes 16-31: row M=lane-16, K 16-31 in v0-7
// One contiguous 16-half chunk per lane -> merges into 2x b128 loads.
__device__ __forceinline__ v16h load_a_frag(const f16* __restrict__ A, int lda,
                                            int row0, int k0) {
  int lane = threadIdx.x & 31;
  int m = lane & 15, kh = lane >> 4;
  const f16* p = A + (long)(row0 + m) * lda + (k0 + 16 * kh);
  v16h a;
#pragma unroll
  for (int v = 0; v < 8; ++v) {
    int bk = 2 * v;
    a[2 * v]     = p[bk];
    a[2 * v + 1] = p[bk + 1];
  }
  return a;
}

// B fragment from pre-shuffled fragment-major weights:
// layout [kt][nt][lane][16] f16 ; element e of a lane's v16h holds
// W[kt*32 + 16*(lane>>4) + e][nt*16 + (lane&15)].
__device__ __forceinline__ v16h load_b_packed(const f16* __restrict__ Wf,
                                              int kt, int nt) {
  int lane = threadIdx.x & 31;
  return *(const v16h*)(Wf + ((((long)kt * 8 + nt) * 32 + lane) << 4));
}

// ---------------------------------------------------------------------------
// Packing kernels: f32 -> f16 with zero padding.
// ---------------------------------------------------------------------------
__global__ void pack_x_f16(const float* __restrict__ src, f16* __restrict__ dst,
                           long M, int K, int Kpad) {
  long idx = (long)blockIdx.x * blockDim.x + threadIdx.x;
  if (idx >= M * Kpad) return;
  int  k = (int)(idx % Kpad);
  long m = idx / Kpad;
  dst[idx] = (f16)((k < K) ? src[m * K + k] : 0.f);
}

// Weights [Krows,128] f32 -> WMMA fragment-major f16 [Kpad/32][8][32][16].
__global__ void pack_w_frag(const float* __restrict__ src, f16* __restrict__ dst,
                            int Krows, int Kpad) {
  int idx = blockIdx.x * blockDim.x + threadIdx.x;
  if (idx >= Kpad * 128) return;
  int e    = idx & 15;
  int lane = (idx >> 4) & 31;
  int nt   = (idx >> 9) & 7;
  int kt   = idx >> 12;
  int K    = kt * 32 + 16 * (lane >> 4) + e;
  int col  = nt * 16 + (lane & 15);
  dst[idx] = (f16)((K < Krows) ? src[K * 128 + col] : 0.f);
}

// ---------------------------------------------------------------------------
// Generic WMMA GEMM: Y[M,128] = act(A[M,Kpad] @ W[Kpad,128] + bias)
// grid.x = M/16, block = 256 (8 waves, wave w owns cols [16w,16w+16)).
// ---------------------------------------------------------------------------
__global__ __launch_bounds__(256) void gemm16(
    const f16* __restrict__ A, int lda, const f16* __restrict__ Wf, int Kpad,
    const float* __restrict__ bias, int relu,
    f16* __restrict__ outh, long ldoh, float* __restrict__ outf, long ldof) {
  int w = threadIdx.x >> 5, lane = threadIdx.x & 31;
  int kh = lane >> 4;
  int colg = w * 16 + (lane & 15);
  int row0 = blockIdx.x * 16;
  v8f acc = {};
#pragma unroll 4
  for (int k0 = 0; k0 < Kpad; k0 += 32) {
    v16h a = load_a_frag(A, lda, row0, k0);
    v16h b = load_b_packed(Wf, k0 >> 5, w);
    acc = WMMA(a, b, acc);
  }
  float bc = bias ? bias[colg] : 0.f;
#pragma unroll
  for (int v = 0; v < 8; ++v) {
    float x = acc[v] + bc;
    if (relu) x = fmaxf(x, 0.f);
    long row = row0 + v + 8 * kh;
    if (outh) outh[row * ldoh + colg] = (f16)x;
    if (outf) outf[row * ldof + colg] = x;
  }
}

// ---------------------------------------------------------------------------
// PairToAtom fused: one block per (b,i) -> 64 pair rows, 2 MLP layers, then
// sum over j.  a1 written as f16 into cat_a[:,128:256].
// ---------------------------------------------------------------------------
__global__ __launch_bounds__(256) void p2a_fused(
    const f16* __restrict__ Xp,                 // [B*N*N, 32] padded pair_x
    const f16* __restrict__ W1f,                // [32,128] frag-major
    const float* __restrict__ b1,
    const f16* __restrict__ W2f,                // [128,128] frag-major
    const float* __restrict__ b2,
    f16* __restrict__ cat_a)                    // [4096,256], cols 128..255
{
  __shared__ f16 h[64][128];
  int bi = blockIdx.x;              // b*64 + i ; pair rows bi*64 .. bi*64+63
  int r0 = bi * 64;
  int w = threadIdx.x >> 5, lane = threadIdx.x & 31;
  int kh = lane >> 4;
  int colg = w * 16 + (lane & 15);
  float bc1 = b1[colg];
  // Layer 1: K=32 (14 real + zero pad), 4 M-tiles of 16 rows.
  v16h b1f = load_b_packed(W1f, 0, w);
#pragma unroll
  for (int mt = 0; mt < 4; ++mt) {
    v16h a = load_a_frag(Xp, 32, r0 + mt * 16, 0);
    v8f acc = {};
    acc = WMMA(a, b1f, acc);
#pragma unroll
    for (int v = 0; v < 8; ++v)
      h[mt * 16 + v + 8 * kh][colg] = (f16)fmaxf(acc[v] + bc1, 0.f);
  }
  __syncthreads();
  // Layer 2 + reduce over the 64 rows (axis j).  Hoist the 4 B fragments.
  v16h b2f[4];
#pragma unroll
  for (int kt = 0; kt < 4; ++kt) b2f[kt] = load_b_packed(W2f, kt, w);
  float bc2 = b2[colg];
  float s = 0.f;
#pragma unroll
  for (int mt = 0; mt < 4; ++mt) {
    v8f acc = {};
#pragma unroll
    for (int kt = 0; kt < 4; ++kt) {
      v16h a = load_a_frag(&h[0][0], 128, mt * 16, kt * 32);
      acc = WMMA(a, b2f[kt], acc);
    }
#pragma unroll
    for (int v = 0; v < 8; ++v) s += fmaxf(acc[v] + bc2, 0.f);
  }
  s += __shfl_xor(s, 16, 32);       // lanes n and n+16 hold same column
  if (lane < 16) cat_a[(long)bi * 256 + 128 + colg] = (f16)s;
}

// ---------------------------------------------------------------------------
// Pair mega-kernel: one block per 32 pair rows (2 M-tiles per wave).
//  h1a = relu(U[i]+V[j]+b)  h1b = relu(U[j]+V[i]+b)        (concat trick)
//  p0  = relu(h1a@W + b) + relu(h1b@W + b)
//  p1  = relu(relu(pair_x@Wp2p1+b)@Wp2p2+b)
//  out = relu(relu(p0@Wpl1a + p1@Wpl1b + b)@Wpl2 + b)
// ---------------------------------------------------------------------------
__global__ __launch_bounds__(256) void pair_fused(
    const f16* __restrict__ Xp,                  // [B*N*N,32] padded pair_x
    const float* __restrict__ U,                 // [4096,128] atom_x @ W1_top
    const float* __restrict__ V,                 // [4096,128] atom_x @ W1_bot
    const float* __restrict__ ba2p1,
    const f16* __restrict__ Wa2p2, const float* __restrict__ ba2p2,
    const f16* __restrict__ Wp2p1, const float* __restrict__ bp2p1,
    const f16* __restrict__ Wp2p2, const float* __restrict__ bp2p2,
    const f16* __restrict__ Wpl1,  const float* __restrict__ bpl1, // [256,128]
    const f16* __restrict__ Wpl2,  const float* __restrict__ bpl2,
    float* __restrict__ out)                     // next_pair [B*N*N,128]
{
  __shared__ f16 bufA[32][128];   // h1a, later h3
  __shared__ f16 bufB[32][128];   // h1b, later p2p hidden
  __shared__ f16 bufP0[32][128];
  __shared__ f16 bufP1[32][128];

  long r0 = (long)blockIdx.x * 32;       // first pair row (32 | 64 => one i)
  int  ai = (int)(r0 >> 6);              // b*64 + i (constant over the tile)
  int  j0 = (int)(r0 & 63);              // 0 or 32
  int  bbase = ai & ~63;                 // b*64
  int w = threadIdx.x >> 5, lane = threadIdx.x & 31;
  int kh = lane >> 4;
  int colg = w * 16 + (lane & 15);

  // Stage A: layer-1 of AtomToPair via the concat trick (float4 vectorized).
  // 32 rows x 32 col-groups of 4.
  for (int idx = threadIdx.x; idx < 32 * 32; idx += 256) {
    int m = idx >> 5, cg = (idx & 31) * 4;
    int jr = bbase + j0 + m;
    float4 bb = *(const float4*)(ba2p1 + cg);
    float4 ui = *(const float4*)(U + (long)ai * 128 + cg);
    float4 vi = *(const float4*)(V + (long)ai * 128 + cg);
    float4 uj = *(const float4*)(U + (long)jr * 128 + cg);
    float4 vj = *(const float4*)(V + (long)jr * 128 + cg);
    v4h ha, hb;
    ha[0] = (f16)fmaxf(ui.x + vj.x + bb.x, 0.f);
    ha[1] = (f16)fmaxf(ui.y + vj.y + bb.y, 0.f);
    ha[2] = (f16)fmaxf(ui.z + vj.z + bb.z, 0.f);
    ha[3] = (f16)fmaxf(ui.w + vj.w + bb.w, 0.f);
    hb[0] = (f16)fmaxf(uj.x + vi.x + bb.x, 0.f);
    hb[1] = (f16)fmaxf(uj.y + vi.y + bb.y, 0.f);
    hb[2] = (f16)fmaxf(uj.z + vi.z + bb.z, 0.f);
    hb[3] = (f16)fmaxf(uj.w + vi.w + bb.w, 0.f);
    *(v4h*)&bufA[m][cg] = ha;
    *(v4h*)&bufB[m][cg] = hb;
  }
  __syncthreads();

  // Stage B: p0 = relu(h1a@W2+b) + relu(h1b@W2+b); each B-frag feeds 4 WMMAs.
  v8f acc1[2] = {}, acc2[2] = {};
#pragma unroll
  for (int kt = 0; kt < 4; ++kt) {
    v16h b = load_b_packed(Wa2p2, kt, w);
#pragma unroll
    for (int mt = 0; mt < 2; ++mt) {
      v16h a1 = load_a_frag(&bufA[0][0], 128, mt * 16, kt * 32);
      acc1[mt] = WMMA(a1, b, acc1[mt]);
      v16h a2 = load_a_frag(&bufB[0][0], 128, mt * 16, kt * 32);
      acc2[mt] = WMMA(a2, b, acc2[mt]);
    }
  }
  // Stage C1 compute (reads only global): h2 = relu(pair_x@Wp2p1+b)
  v8f acc3[2] = {};
  {
    v16h b = load_b_packed(Wp2p1, 0, w);
#pragma unroll
    for (int mt = 0; mt < 2; ++mt) {
      v16h a = load_a_frag(Xp, 32, (int)r0 + mt * 16, 0);
      acc3[mt] = WMMA(a, b, acc3[mt]);
    }
  }
  __syncthreads();                       // all reads of bufA/bufB complete
  {
    float bc = ba2p2[colg], bc3 = bp2p1[colg];
#pragma unroll
    for (int mt = 0; mt < 2; ++mt)
#pragma unroll
      for (int v = 0; v < 8; ++v) {
        int m = mt * 16 + v + 8 * kh;
        bufP0[m][colg] = (f16)(fmaxf(acc1[mt][v] + bc, 0.f) +
                               fmaxf(acc2[mt][v] + bc, 0.f));
        bufB[m][colg]  = (f16)fmaxf(acc3[mt][v] + bc3, 0.f);
      }
  }
  __syncthreads();

  // Stage C2: p1 = relu(h2@Wp2p2+b)
  {
    v8f acc[2] = {};
#pragma unroll
    for (int kt = 0; kt < 4; ++kt) {
      v16h b = load_b_packed(Wp2p2, kt, w);
#pragma unroll
      for (int mt = 0; mt < 2; ++mt) {
        v16h a = load_a_frag(&bufB[0][0], 128, mt * 16, kt * 32);
        acc[mt] = WMMA(a, b, acc[mt]);
      }
    }
    float bc = bp2p2[colg];
#pragma unroll
    for (int mt = 0; mt < 2; ++mt)
#pragma unroll
      for (int v = 0; v < 8; ++v)
        bufP1[mt * 16 + v + 8 * kh][colg] = (f16)fmaxf(acc[mt][v] + bc, 0.f);
  }
  __syncthreads();

  // Stage D: h3 = relu(p0@Wpl1[0:128] + p1@Wpl1[128:256] + b)  (split concat)
  {
    v8f acc[2] = {};
#pragma unroll
    for (int kt = 0; kt < 4; ++kt) {
      v16h b = load_b_packed(Wpl1, kt, w);
#pragma unroll
      for (int mt = 0; mt < 2; ++mt) {
        v16h a = load_a_frag(&bufP0[0][0], 128, mt * 16, kt * 32);
        acc[mt] = WMMA(a, b, acc[mt]);
      }
    }
#pragma unroll
    for (int kt = 0; kt < 4; ++kt) {
      v16h b = load_b_packed(Wpl1, 4 + kt, w);   // rows 128..255
#pragma unroll
      for (int mt = 0; mt < 2; ++mt) {
        v16h a = load_a_frag(&bufP1[0][0], 128, mt * 16, kt * 32);
        acc[mt] = WMMA(a, b, acc[mt]);
      }
    }
    float bc = bpl1[colg];
    __syncthreads();
#pragma unroll
    for (int mt = 0; mt < 2; ++mt)
#pragma unroll
      for (int v = 0; v < 8; ++v)
        bufA[mt * 16 + v + 8 * kh][colg] = (f16)fmaxf(acc[mt][v] + bc, 0.f);
  }
  __syncthreads();

  // Stage E: next_pair = relu(h3@Wpl2 + b)
  {
    v8f acc[2] = {};
#pragma unroll
    for (int kt = 0; kt < 4; ++kt) {
      v16h b = load_b_packed(Wpl2, kt, w);
#pragma unroll
      for (int mt = 0; mt < 2; ++mt) {
        v16h a = load_a_frag(&bufA[0][0], 128, mt * 16, kt * 32);
        acc[mt] = WMMA(a, b, acc[mt]);
      }
    }
    float bc = bpl2[colg];
#pragma unroll
    for (int mt = 0; mt < 2; ++mt)
#pragma unroll
      for (int v = 0; v < 8; ++v)
        out[(r0 + mt * 16 + v + 8 * kh) * 128 + colg] =
            fmaxf(acc[mt][v] + bc, 0.f);
  }
}

// ---------------------------------------------------------------------------
// Host launcher
// ---------------------------------------------------------------------------
extern "C" void kernel_launch(void* const* d_in, const int* in_sizes, int n_in,
                              void* d_out, int out_size, void* d_ws, size_t ws_size,
                              hipStream_t stream) {
  (void)in_sizes; (void)n_in; (void)out_size; (void)ws_size;
  const int Bn = 64, Nn = 64, C = 128;
  const int BN = Bn * Nn;          // 4096 atom rows
  const long NP = (long)BN * Nn;   // 262144 pair rows

  const float* atom_x = (const float*)d_in[0];
  const float* pair_x = (const float*)d_in[1];
  const float* Wa2a1 = (const float*)d_in[2];  const float* ba2a1 = (const float*)d_in[3];
  const float* Wa2a2 = (const float*)d_in[4];  const float* ba2a2 = (const float*)d_in[5];
  const float* Wp2a1 = (const float*)d_in[6];  const float* bp2a1 = (const float*)d_in[7];
  const float* Wp2a2 = (const float*)d_in[8];  const float* bp2a2 = (const float*)d_in[9];
  const float* Wa2p1 = (const float*)d_in[10]; const float* ba2p1 = (const float*)d_in[11];
  const float* Wa2p2 = (const float*)d_in[12]; const float* ba2p2 = (const float*)d_in[13];
  const float* Wp2p1 = (const float*)d_in[14]; const float* bp2p1 = (const float*)d_in[15];
  const float* Wp2p2 = (const float*)d_in[16]; const float* bp2p2 = (const float*)d_in[17];
  const float* Wat1  = (const float*)d_in[18]; const float* bat1  = (const float*)d_in[19];
  const float* Wat2  = (const float*)d_in[20]; const float* bat2  = (const float*)d_in[21];
  const float* Wpl1  = (const float*)d_in[22]; const float* bpl1  = (const float*)d_in[23];
  const float* Wpl2  = (const float*)d_in[24]; const float* bpl2  = (const float*)d_in[25];

  float* out_atom = (float*)d_out;                     // [4096,128]
  float* out_pair = (float*)d_out + (long)BN * C;      // [262144,128]

  // Workspace carve-out (all fully overwritten every call).
  size_t off = 0;
  auto wsa = [&](size_t bytes) -> char* {
    char* p = (char*)d_ws + off;
    off += (bytes + 255) & ~(size_t)255;
    return p;
  };
  f16*   atomf  = (f16*)wsa((size_t)BN * 96 * 2);      // atom_x padded 75->96
  f16*   pairf  = (f16*)wsa((size_t)NP * 32 * 2);      // pair_x padded 14->32
  f16*   wA2A1  = (f16*)wsa(96  * 128 * 2);
  f16*   wA2A2  = (f16*)wsa(128 * 128 * 2);
  f16*   wP2A1  = (f16*)wsa(32  * 128 * 2);
  f16*   wP2A2  = (f16*)wsa(128 * 128 * 2);
  f16*   wA2PT  = (f16*)wsa(96  * 128 * 2);            // a2p W1 rows 0..74
  f16*   wA2PB  = (f16*)wsa(96  * 128 * 2);            // a2p W1 rows 75..149
  f16*   wA2P2  = (f16*)wsa(128 * 128 * 2);
  f16*   wP2P1  = (f16*)wsa(32  * 128 * 2);
  f16*   wP2P2  = (f16*)wsa(128 * 128 * 2);
  f16*   wAT1   = (f16*)wsa(256 * 128 * 2);
  f16*   wAT2   = (f16*)wsa(128 * 128 * 2);
  f16*   wPL1   = (f16*)wsa(256 * 128 * 2);
  f16*   wPL2   = (f16*)wsa(128 * 128 * 2);
  float* Ubuf   = (float*)wsa((size_t)BN * C * 4);
  float* Vbuf   = (float*)wsa((size_t)BN * C * 4);
  f16*   hA2A   = (f16*)wsa((size_t)BN * C * 2);
  f16*   catA   = (f16*)wsa((size_t)BN * 2 * C * 2);   // [4096,256]
  f16*   hATOM  = (f16*)wsa((size_t)BN * C * 2);

  // --- Phase 0: pack activations and weights (weights -> fragment-major) ---
  {
    long n1 = (long)BN * 96;
    pack_x_f16<<<(unsigned)((n1 + 255) / 256), 256, 0, stream>>>(atom_x, atomf, BN, 75, 96);
    long n2 = NP * 32;
    pack_x_f16<<<(unsigned)((n2 + 255) / 256), 256, 0, stream>>>(pair_x, pairf, NP, 14, 32);
  }
  pack_w_frag<<<96  * 128 / 256, 256, 0, stream>>>(Wa2a1, wA2A1, 75, 96);
  pack_w_frag<<<128 * 128 / 256, 256, 0, stream>>>(Wa2a2, wA2A2, 128, 128);
  pack_w_frag<<<32  * 128 / 256, 256, 0, stream>>>(Wp2a1, wP2A1, 14, 32);
  pack_w_frag<<<128 * 128 / 256, 256, 0, stream>>>(Wp2a2, wP2A2, 128, 128);
  pack_w_frag<<<96  * 128 / 256, 256, 0, stream>>>(Wa2p1,            wA2PT, 75, 96);
  pack_w_frag<<<96  * 128 / 256, 256, 0, stream>>>(Wa2p1 + 75 * 128, wA2PB, 75, 96);
  pack_w_frag<<<128 * 128 / 256, 256, 0, stream>>>(Wa2p2, wA2P2, 128, 128);
  pack_w_frag<<<32  * 128 / 256, 256, 0, stream>>>(Wp2p1, wP2P1, 14, 32);
  pack_w_frag<<<128 * 128 / 256, 256, 0, stream>>>(Wp2p2, wP2P2, 128, 128);
  pack_w_frag<<<256 * 128 / 256, 256, 0, stream>>>(Wat1, wAT1, 256, 256);
  pack_w_frag<<<128 * 128 / 256, 256, 0, stream>>>(Wat2, wAT2, 128, 128);
  pack_w_frag<<<256 * 128 / 256, 256, 0, stream>>>(Wpl1, wPL1, 256, 256);
  pack_w_frag<<<128 * 128 / 256, 256, 0, stream>>>(Wpl2, wPL2, 128, 128);

  // --- Phase 1: atom-side GEMMs (4096 rows) ---
  gemm16<<<BN / 16, 256, 0, stream>>>(atomf, 96, wA2A1, 96, ba2a1, 1,
                                      hA2A, 128, nullptr, 0);
  gemm16<<<BN / 16, 256, 0, stream>>>(hA2A, 128, wA2A2, 128, ba2a2, 1,
                                      catA, 256, nullptr, 0);
  gemm16<<<BN / 16, 256, 0, stream>>>(atomf, 96, wA2PT, 96, nullptr, 0,
                                      nullptr, 0, Ubuf, 128);
  gemm16<<<BN / 16, 256, 0, stream>>>(atomf, 96, wA2PB, 96, nullptr, 0,
                                      nullptr, 0, Vbuf, 128);

  // --- Phase 2: pair_to_atom fused (2 layers + j-reduction) -> cat_a[:,128:256]
  p2a_fused<<<BN, 256, 0, stream>>>(pairf, wP2A1, bp2a1, wP2A2, bp2a2, catA);

  // --- Phase 3: atom_layer on concat(a0, a1) ---
  gemm16<<<BN / 16, 256, 0, stream>>>(catA, 256, wAT1, 256, bat1, 1,
                                      hATOM, 128, nullptr, 0);
  gemm16<<<BN / 16, 256, 0, stream>>>(hATOM, 128, wAT2, 128, bat2, 1,
                                      nullptr, 0, out_atom, 128);

  // --- Phase 4: fused pair branch -> next_pair (32 rows/block) ---
  pair_fused<<<(unsigned)(NP / 32), 256, 0, stream>>>(
      pairf, Ubuf, Vbuf, ba2p1, wA2P2, ba2p2, wP2P1, bp2p1, wP2P2, bp2p2,
      wPL1, bpl1, wPL2, bpl2, out_pair);
}
